// DoubleAttention_90494960926888
// MI455X (gfx1250) — matrix operand order repaired and checked by reading, not compiled, not run on hardware
//
#include <hip/hip_runtime.h>
#include <cstdint>

// Dims fixed by the reference: B=4, L=2048, H=8, E=64, NUM_BUCKETS=32, MAX_DISTANCE=128
#define B_DIM 4
#define L_DIM 2048
#define H_DIM 8
#define E_DIM 64
#define NEG_INF (-1e30f)

typedef __attribute__((ext_vector_type(16))) _Float16 v16h;
typedef __attribute__((ext_vector_type(8)))  _Float16 v8h;
typedef __attribute__((ext_vector_type(8)))  float    v8f;

static __device__ __forceinline__ v8f wmma_f16(v16h a, v16h b, v8f c) {
  // D = A(16x32 f16) * B(32x16 f16) + C(16x16 f32)
  return __builtin_amdgcn_wmma_f32_16x16x32_f16(false, a, false, b, (short)0, c,
                                                false, false);
}

// convert 8 contiguous f32 -> 8 f16 elements of a v16h, starting at `at`
static __device__ __forceinline__ void cvt8(const float* __restrict__ p, v16h& d, int at) {
  float4 f0 = ((const float4*)p)[0];
  float4 f1 = ((const float4*)p)[1];
  d[at + 0] = (_Float16)f0.x; d[at + 1] = (_Float16)f0.y;
  d[at + 2] = (_Float16)f0.z; d[at + 3] = (_Float16)f0.w;
  d[at + 4] = (_Float16)f1.x; d[at + 5] = (_Float16)f1.y;
  d[at + 6] = (_Float16)f1.z; d[at + 7] = (_Float16)f1.w;
}

// B-operand layout: per lane (half h) k = [16h, 16h+16) contiguous
static __device__ __forceinline__ v16h load_b_row(const float* __restrict__ p) {
  v16h r; cvt8(p, r, 0); cvt8(p + 8, r, 8); return r;
}

static __device__ __forceinline__ float rmax16(float x) {
  x = fmaxf(x, __shfl_xor(x, 1, 32));
  x = fmaxf(x, __shfl_xor(x, 2, 32));
  x = fmaxf(x, __shfl_xor(x, 4, 32));
  x = fmaxf(x, __shfl_xor(x, 8, 32));
  return x;
}
static __device__ __forceinline__ float rsum16(float x) {
  x += __shfl_xor(x, 1, 32);
  x += __shfl_xor(x, 2, 32);
  x += __shfl_xor(x, 4, 32);
  x += __shfl_xor(x, 8, 32);
  return x;
}

// ---------------------------------------------------------------------------
// Pre-pass: V [B,L,H,E] f32  ->  Vt [B*H][E][L] f16 (transposed per (b,h))
// ---------------------------------------------------------------------------
__global__ __launch_bounds__(256) void vtrans_kernel(const float* __restrict__ V,
                                                     _Float16* __restrict__ Vt) {
  __shared__ _Float16 tile[32][33];
  const int s0 = blockIdx.x * 32;         // L/32 tiles
  const int e0 = blockIdx.y * 32;         // E/32 tiles
  const int bh = blockIdx.z;              // B*H
  const int b = bh / H_DIM, h = bh % H_DIM;
  const int tx = threadIdx.x;             // 0..31
  const int ty = threadIdx.y;             // 0..7
#pragma unroll
  for (int i = 0; i < 4; ++i) {
    int s = s0 + ty + i * 8;
    tile[ty + i * 8][tx] =
        (_Float16)V[(((size_t)b * L_DIM + s) * H_DIM + h) * E_DIM + e0 + tx];
  }
  __syncthreads();
#pragma unroll
  for (int i = 0; i < 4; ++i) {
    int e = e0 + ty + i * 8;
    Vt[((size_t)bh * E_DIM + e) * L_DIM + s0 + tx] = tile[tx][ty + i * 8];
  }
}

// ---------------------------------------------------------------------------
// Flash-attention-2 style kernel. 4 waves / WG, each wave owns 16 query rows.
// 64 keys per main-loop iteration -> 16 WMMAs per iteration.
// ---------------------------------------------------------------------------
__global__ __launch_bounds__(128) void fa2_kernel(const float* __restrict__ Q,
                                                  const float* __restrict__ K,
                                                  const _Float16* __restrict__ Vt,
                                                  const float* __restrict__ bias_table,
                                                  float* __restrict__ out) {
  __shared__ float bias_lds[L_DIM];                       // bias per distance d
  __shared__ __align__(32) _Float16 p_lds[4][16 * 64];    // per-wave P staging

  const int qblock = blockIdx.x % (L_DIM / 64);
  const int bh     = blockIdx.x / (L_DIM / 64);
  const int b      = bh / H_DIM;
  const int head   = bh % H_DIM;
  const int tid    = threadIdx.x;

  // ---- per-WG bias lookup table: bias_lds[d] = bias_table[bucket(d)*8 + head]
  const float inv16   = 1.0f / 16.0f;
  const float invLog8 = 16.0f / 2.0794415416798357f;      // 16 / ln(8)
#pragma unroll 1
  for (int d = tid; d < L_DIM; d += 128) {
    int bkt;
    if (d < 16) {
      bkt = d;
    } else {
      bkt = 16 + (int)(logf((float)d * inv16) * invLog8);
      if (bkt > 31) bkt = 31;
    }
    bias_lds[d] = bias_table[bkt * 8 + head];
  }
  __syncthreads();

  const int wave  = tid >> 5;
  const int lane  = tid & 31;
  const int half  = lane >> 4;   // k-half selector for A/B operands
  const int l16   = lane & 15;   // row (A) / column (B,D) index
  const int qtile = qblock * 64 + wave * 16;

  // ---- load Q tile in WMMA A-layout (per lane: k = [16h, 16h+16) contiguous)
  const float* qrow = Q + (((size_t)b * L_DIM + qtile + l16) * H_DIM + head) * E_DIM;
  v16h qa, qb;
  cvt8(qrow + 16 * half,      qa, 0);
  cvt8(qrow + 8 + 16 * half,  qa, 8);
  cvt8(qrow + 32 + 16 * half, qb, 0);
  cvt8(qrow + 40 + 16 * half, qb, 8);

  v8f o0 = {}, o1 = {}, o2 = {}, o3 = {};
  float mI[8], lpart[8];
#pragma unroll
  for (int r = 0; r < 8; ++r) { mI[r] = NEG_INF; lpart[r] = 0.0f; }

  const float* kbase = K + ((size_t)b * L_DIM * H_DIM + head) * E_DIM;
  const _Float16* vtb = Vt + (size_t)bh * E_DIM * L_DIM;
  _Float16* pl = &p_lds[wave][0];

  const float scl  = 0.125f;                 // 1/sqrt(64)
  const int   send = qtile + 16;

  for (int sbase = 0; sbase < send; sbase += 64) {
    // prefetch next block's K rows (one lane per row; speculative, range-guarded)
    if (sbase + 64 < send) {
      __builtin_prefetch(kbase + (size_t)(sbase + 64 + lane) * H_DIM * E_DIM, 0, 1);
    }

    // ---- S = Q * K^T for 64 keys (four 16-wide N tiles)
    v8f s[4];
#pragma unroll
    for (int t = 0; t < 4; ++t) {
      const float* krow = kbase + (size_t)(sbase + 16 * t + l16) * H_DIM * E_DIM;
      v16h k0 = load_b_row(krow + 16 * half);
      v16h k1 = load_b_row(krow + 32 + 16 * half);
      v8f acc = {};
      acc = wmma_f16(qa, k0, acc);
      acc = wmma_f16(qb, k1, acc);
      s[t] = acc;
    }

    // ---- bias + causal mask + online softmax (row m = r + 8*half, col n = l16)
#pragma unroll
    for (int r = 0; r < 8; ++r) {
      const int m  = qtile + r + 8 * half;
      const int d0 = m - (sbase + l16);
      float x[4];
#pragma unroll
      for (int t = 0; t < 4; ++t) {
        const int dt = d0 - 16 * t;
        const float bt = bias_lds[dt < 0 ? 0 : dt];
        x[t] = (dt >= 0) ? (s[t][r] * scl + bt) : NEG_INF;
      }

      float mnew = fmaxf(fmaxf(x[0], x[1]), fmaxf(x[2], x[3]));
      mnew = fmaxf(mI[r], rmax16(mnew));
      const float alpha = __expf(mI[r] - mnew);
      mI[r] = mnew;

      float p0 = __expf(x[0] - mnew);
      float p1 = __expf(x[1] - mnew);
      float p2 = __expf(x[2] - mnew);
      float p3 = __expf(x[3] - mnew);
      // deferred row-sum: every lane of the half applies the same alpha chain,
      // so lane-local partial sums reduce correctly once at the end.
      lpart[r] = lpart[r] * alpha + ((p0 + p1) + (p2 + p3));

      o0[r] *= alpha; o1[r] *= alpha; o2[r] *= alpha; o3[r] *= alpha;

      const int mloc = (r + 8 * half) * 64 + l16;
      pl[mloc]      = (_Float16)p0;
      pl[mloc + 16] = (_Float16)p1;
      pl[mloc + 32] = (_Float16)p2;
      pl[mloc + 48] = (_Float16)p3;
    }

    // same-wave DS ops are in-order; wait to be safe before re-reading P
    asm volatile("s_wait_dscnt 0" ::: "memory");

    // ---- O += P * V  over the 64-key block, in two K=32 chunks
#pragma unroll
    for (int c = 0; c < 2; ++c) {
      // A2 = P chunk (16x32), A-layout contiguous [32c+16h,+16)
      v8h pc0 = *(const v8h*)(pl + l16 * 64 + 32 * c + 16 * half);
      v8h pc1 = *(const v8h*)(pl + l16 * 64 + 32 * c + 8 + 16 * half);
      v16h a2;
#pragma unroll
      for (int i = 0; i < 8; ++i) { a2[i] = pc0[i]; a2[8 + i] = pc1[i]; }

      const int soff = sbase + 32 * c + 16 * half;
      v16h b2;
      b2 = *(const v16h*)(vtb + (size_t)(0 * 16 + l16) * L_DIM + soff);
      o0 = wmma_f16(a2, b2, o0);
      b2 = *(const v16h*)(vtb + (size_t)(1 * 16 + l16) * L_DIM + soff);
      o1 = wmma_f16(a2, b2, o1);
      b2 = *(const v16h*)(vtb + (size_t)(2 * 16 + l16) * L_DIM + soff);
      o2 = wmma_f16(a2, b2, o2);
      b2 = *(const v16h*)(vtb + (size_t)(3 * 16 + l16) * L_DIM + soff);
      o3 = wmma_f16(a2, b2, o3);
    }
  }

  // ---- epilogue: final row-sum reduction, then out[b, m, head, e] = O/l
#pragma unroll
  for (int r = 0; r < 8; ++r) {
    const float inv = 1.0f / rsum16(lpart[r]);
    const int m = qtile + r + 8 * half;
    float* orow = out + (((size_t)b * L_DIM + m) * H_DIM + head) * E_DIM + l16;
    orow[0]  = o0[r] * inv;
    orow[16] = o1[r] * inv;
    orow[32] = o2[r] * inv;
    orow[48] = o3[r] * inv;
  }
}

extern "C" void kernel_launch(void* const* d_in, const int* in_sizes, int n_in,
                              void* d_out, int out_size, void* d_ws, size_t ws_size,
                              hipStream_t stream) {
  const float* Q  = (const float*)d_in[0];
  const float* K  = (const float*)d_in[1];
  const float* V  = (const float*)d_in[2];
  const float* bt = (const float*)d_in[3];
  float* out = (float*)d_out;

  // workspace: transposed f16 V, B*H*E*L f16 = 8 MB
  _Float16* Vt = (_Float16*)d_ws;

  vtrans_kernel<<<dim3(L_DIM / 32, E_DIM / 32, B_DIM * H_DIM), dim3(32, 8), 0, stream>>>(V, Vt);

  fa2_kernel<<<dim3((L_DIM / 64) * B_DIM * H_DIM), dim3(128), 0, stream>>>(Q, K, Vt, bt, out);
}
